// BasicTransformerBlock_34488587387355
// MI455X (gfx1250) — matrix operand order repaired and checked by reading, not compiled, not run on hardware
//
#include <hip/hip_runtime.h>
#include <hip/hip_bf16.h>
#include <math.h>
#include <stdint.h>

// ---------------------------------------------------------------------------
// Types for CDNA5 WMMA (wave32): 16x16x32 bf16 -> f32
// ---------------------------------------------------------------------------
typedef __attribute__((ext_vector_type(16))) __bf16 v16bf;
typedef __attribute__((ext_vector_type(8)))  float  v8f;
typedef __attribute__((ext_vector_type(4)))  unsigned int v4u;
typedef __attribute__((ext_vector_type(4)))  unsigned int u32x4;
typedef __attribute__((ext_vector_type(8)))  int i32x8;
typedef __attribute__((ext_vector_type(4)))  int i32x4;

#define WMMA_BF16(a, b, c) \
  __builtin_amdgcn_wmma_f32_16x16x32_bf16(false, (a), false, (b), (short)0, (c), false, false)

// Tensor Data Mover path (gfx1250): guarded so compile stays green on any
// toolchain; fallback is plain global loads. This toolchain uses the 6-arg
// builtin form (u32x4, i32x8, i32x4, i32x4, i32x8, i32 cpol).
#if defined(__AMDGCN__) && __has_builtin(__builtin_amdgcn_tensor_load_to_lds) && \
    __has_builtin(__builtin_amdgcn_s_wait_tensorcnt)
#define USE_TDM 1
#else
#define USE_TDM 0
#endif

union FragU { v4u u[2]; v16bf v; __bf16 e[16]; };
union H8   { v4u u;     __bf16 e[8]; };

__device__ inline v16bf zero_frag() {
  FragU t;
  v4u z = {0u, 0u, 0u, 0u};
  t.u[0] = z; t.u[1] = z;
  return t.v;
}

// A-fragment (16x32 bf16, row-major source, ISA layout):
// lane<16: row M=lane, elems 0..7 = K k0..k0+7, elems 8..15 = K k0+16..k0+23
// lane>=16: row M=lane-16, K offset +8 / +24. Two contiguous 16B loads.
__device__ inline v16bf load_a_frag(const __bf16* __restrict__ A, int lda,
                                    int M, int m0, int k0) {
  int lane = threadIdx.x & 31;
  int r16  = lane & 15;
  int hi   = lane >> 4;
  int row  = m0 + r16;
  FragU t;
  if (row < M) {
    const __bf16* p = A + (size_t)row * lda + k0 + hi * 8;
    t.u[0] = *(const v4u*)(p);
    t.u[1] = *(const v4u*)(p + 16);
  } else {
    v4u z = {0u, 0u, 0u, 0u};
    t.u[0] = z; t.u[1] = z;
  }
  return t.v;
}

// A-fragment from a packed 16x32 LDS tile (row r contiguous 32 elems)
__device__ inline v16bf load_a_frag_lds(const __bf16* tile) {
  int lane = threadIdx.x & 31;
  int r16  = lane & 15;
  int hi   = lane >> 4;
  const __bf16* p = tile + r16 * 32 + hi * 8;
  FragU t;
  t.u[0] = *(const v4u*)(p);
  t.u[1] = *(const v4u*)(p + 16);
  return t.v;
}

// B-fragment from two 8-elem runs at p and p+16 (col-major staged: [n][k]):
// lane<16: col n=lane, K 0..7,16..23 ; lane>=16: col n=lane-16, K 8..15,24..31
__device__ inline v16bf load_b16(const __bf16* p) {
  FragU t;
  t.u[0] = *(const v4u*)(p);
  t.u[1] = *(const v4u*)(p + 16);
  return t.v;
}

#if USE_TDM
// Issue TENSOR_LOAD_TO_LDS of a 16-row x 32-col bf16 tile starting at gaddr
// (row stride = lda elements). rows_left clamps the Y tensor dim so that rows
// past M are hardware-zero-filled (D# OOB rule). 2D tile -> groups 2/3 zero.
__device__ inline void tdm_load_a_tile(const __bf16* gaddr, unsigned lds_off,
                                       int rows_left, int k_left, int lda) {
  unsigned long long ga = (unsigned long long)(uintptr_t)gaddr;
  unsigned td0 = (unsigned)k_left;                       // x extent (elems)
  unsigned td1 = (unsigned)(rows_left > 0 ? rows_left : 0); // y extent (rows)
  u32x4 g0;
  g0[0] = 1u;                                            // count=1, no gather
  g0[1] = lds_off;                                       // LDS byte address
  g0[2] = (unsigned)(ga & 0xffffffffu);                  // global addr lo
  g0[3] = ((unsigned)((ga >> 32) & 0x01ffffffu)) | (2u << 30); // hi | type=2
  i32x8 g1;
  g1[0] = (int)(1u << 16);                               // data_size=1 (2B)
  g1[1] = (int)((td0 & 0xffffu) << 16);                  // tensor_dim0[15:0]
  g1[2] = (int)((td0 >> 16) | ((td1 & 0xffffu) << 16));  // dim0 hi | dim1 lo
  g1[3] = (int)((td1 >> 16) | (32u << 16));              // dim1 hi | tile_dim0=32
  g1[4] = (int)16u;                                      // tile_dim1=16, dim2=0
  g1[5] = (int)(unsigned)lda;                            // dim0_stride lo
  g1[6] = 0;                                             // stride hi | dim1_stride
  g1[7] = 0;
  i32x4 z4 = {0, 0, 0, 0};
  i32x8 z8 = {0, 0, 0, 0, 0, 0, 0, 0};
  __builtin_amdgcn_tensor_load_to_lds(g0, g1, z4, z4, z8, 0);
}
#endif

// ---------------------------------------------------------------------------
// Convert f32 -> bf16
// ---------------------------------------------------------------------------
__global__ void cvt_bf16_kernel(const float* __restrict__ s,
                                __bf16* __restrict__ d, long long n) {
  long long i = (long long)blockIdx.x * blockDim.x + threadIdx.x;
  long long stride = (long long)gridDim.x * blockDim.x;
  for (; i < n; i += stride) d[i] = (__bf16)s[i];
}

// ---------------------------------------------------------------------------
// LayerNorm (row length D) -> bf16
// ---------------------------------------------------------------------------
__global__ __launch_bounds__(256)
void ln_bf16_kernel(const float* __restrict__ x, const float* __restrict__ gm,
                    const float* __restrict__ bt, __bf16* __restrict__ out, int D) {
  __shared__ float rs[256], rq[256];
  int row = blockIdx.x;
  const float* xr = x + (size_t)row * D;
  float s = 0.f, q = 0.f;
  for (int i = threadIdx.x; i < D; i += 256) { float v = xr[i]; s += v; q += v * v; }
  rs[threadIdx.x] = s; rq[threadIdx.x] = q;
  __syncthreads();
  for (int off = 128; off > 0; off >>= 1) {
    if (threadIdx.x < off) {
      rs[threadIdx.x] += rs[threadIdx.x + off];
      rq[threadIdx.x] += rq[threadIdx.x + off];
    }
    __syncthreads();
  }
  float mu  = rs[0] / (float)D;
  float var = rq[0] / (float)D - mu * mu;
  float ri  = rsqrtf(var + 1e-5f);
  __bf16* orow = out + (size_t)row * D;
  for (int i = threadIdx.x; i < D; i += 256)
    orow[i] = (__bf16)((xr[i] - mu) * ri * gm[i] + bt[i]);
}

// ---------------------------------------------------------------------------
// GEMM: C(MxN,f32) = A(MxK,bf16) * B(KxN,bf16) [+bias] [+res] -> f32/bf16
// GEGLU mode (B2 != null): out = (C+bias) * gelu(C2+bias2), written bf16.
// Block: 128 threads = 4 waves, block tile 64(M)x64(N); wave tile 16x64.
// B tile (32 x 64) staged transposed in LDS: bt[n][k].
// A tile: per-wave 16x32, double-buffered via TDM (tensor_load_to_lds) when
// available; TENSORcnt<=1 overlaps next tile DMA with current tile's WMMAs.
// ---------------------------------------------------------------------------
__global__ __launch_bounds__(128)
void gemm_bf16_kernel(const __bf16* __restrict__ A, int lda,
                      const __bf16* __restrict__ B, int ldb,
                      const __bf16* __restrict__ B2,
                      const float* __restrict__ bias,
                      const float* __restrict__ bias2,
                      const float* __restrict__ res,
                      float* __restrict__ outF,
                      __bf16* __restrict__ outB,
                      int ldo, int M, int N, int Kd) {
  __shared__ __bf16 bt [64 * 32];
  __shared__ __bf16 bt2[64 * 32];
#if USE_TDM
  __shared__ __bf16 at[2][4][16 * 32];   // [buffer][wave][packed tile]
#endif
  const int lane = threadIdx.x & 31;
  const int wave = threadIdx.x >> 5;
  const int r16  = lane & 15;
  const int hi   = lane >> 4;
  const int m0   = blockIdx.y * 64 + wave * 16;
  const int n0   = blockIdx.x * 64;
  const bool geglu = (B2 != nullptr);

  v8f acc[4], accg[4];
  for (int t = 0; t < 4; t++)
    for (int i = 0; i < 8; i++) { acc[t][i] = 0.f; accg[t][i] = 0.f; }

#if USE_TDM
  const unsigned lds0 = (unsigned)(uintptr_t)(const void*)&at[0][wave][0];
  const unsigned lds1 = (unsigned)(uintptr_t)(const void*)&at[1][wave][0];
  tdm_load_a_tile(A + (size_t)m0 * lda, lds0, M - m0, Kd, lda);
#endif

  for (int k0 = 0; k0 < Kd; k0 += 32) {
    const int it = k0 >> 5;
    const bool more = (k0 + 32) < Kd;
#if USE_TDM
    if (more)
      tdm_load_a_tile(A + (size_t)m0 * lda + (k0 + 32),
                      ((it + 1) & 1) ? lds1 : lds0, M - m0, Kd - (k0 + 32), lda);
#endif
    __syncthreads();
    // stage B (and B2) transposed: 32 k-rows x 64 n-cols -> bt[n][k]
    for (int idx = threadIdx.x; idx < 256; idx += 128) {
      int kk = idx >> 3;
      int nn = (idx & 7) * 8;
      H8 dv;
      dv.u = *(const v4u*)(B + (size_t)(k0 + kk) * ldb + n0 + nn);
#pragma unroll
      for (int e = 0; e < 8; e++) bt[(nn + e) * 32 + kk] = dv.e[e];
      if (geglu) {
        dv.u = *(const v4u*)(B2 + (size_t)(k0 + kk) * ldb + n0 + nn);
#pragma unroll
        for (int e = 0; e < 8; e++) bt2[(nn + e) * 32 + kk] = dv.e[e];
      }
    }
    if (more)  // prefetch next B tile rows into L2
      __builtin_prefetch(B + (size_t)(k0 + 32 + (threadIdx.x & 31)) * ldb + n0 +
                             ((threadIdx.x >> 5) << 4), 0, 0);
    __syncthreads();

#if USE_TDM
    if (more) __builtin_amdgcn_s_wait_tensorcnt(1);
    else      __builtin_amdgcn_s_wait_tensorcnt(0);
    v16bf a = load_a_frag_lds(&at[it & 1][wave][0]);
#else
    v16bf a = load_a_frag(A, lda, M, m0, k0);
#endif

    v16bf bfr[4];
#pragma unroll
    for (int t = 0; t < 4; t++)
      bfr[t] = load_b16(&bt[(t * 16 + r16) * 32 + hi * 8]);
#pragma unroll
    for (int t = 0; t < 4; t++)
      acc[t] = WMMA_BF16(a, bfr[t], acc[t]);
    if (geglu) {
#pragma unroll
      for (int t = 0; t < 4; t++)
        bfr[t] = load_b16(&bt2[(t * 16 + r16) * 32 + hi * 8]);
#pragma unroll
      for (int t = 0; t < 4; t++)
        accg[t] = WMMA_BF16(a, bfr[t], accg[t]);
    }
  }

  // epilogue. C layout: elem r -> row m0 + r + hi*8, col n0 + t*16 + (lane&15)
#pragma unroll
  for (int t = 0; t < 4; t++) {
    int col = n0 + t * 16 + r16;
#pragma unroll
    for (int r = 0; r < 8; r++) {
      int row = m0 + r + hi * 8;
      if (row < M) {
        float v = acc[t][r];
        if (bias) v += bias[col];
        if (geglu) {
          float g = accg[t][r];
          if (bias2) g += bias2[col];
          v = v * (0.5f * g * (1.0f + erff(g * 0.70710678118654752f)));
        }
        if (res)  v += res[(size_t)row * ldo + col];
        if (outF) outF[(size_t)row * ldo + col] = v;
        if (outB) outB[(size_t)row * ldo + col] = (__bf16)v;
      }
    }
  }
}

// ---------------------------------------------------------------------------
// Flash attention: one wave per (batch, head, 16-query block).
// Q/K/V row stride = 1024 (head-dim 64 contiguous). Online softmax over keys
// processed 32 at a time. S = Q*K^T via 2x wmma (Kdim 64); O += P*V via 4x wmma
// per step (V staged transposed in LDS; P transposed through LDS to A-layout).
// ---------------------------------------------------------------------------
__global__ __launch_bounds__(32)
void flash_attn_kernel(const __bf16* __restrict__ Q,
                       const __bf16* __restrict__ Kp,
                       const __bf16* __restrict__ V,
                       __bf16* __restrict__ O, int nq, int nk) {
  __shared__ __bf16 vt[64 * 32];  // vt[d][k]
  __shared__ float  pp[16 * 32];  // pp[row][key]
  const int lane = threadIdx.x;
  const int r16  = lane & 15;
  const int hi   = lane >> 4;
  const int q0   = blockIdx.x * 16;
  const int h    = blockIdx.y;
  const int b    = blockIdx.z;
  const int LD   = 1024;
  const size_t qbase = (size_t)b * nq * LD + h * 64;
  const size_t kbase = (size_t)b * nk * LD + h * 64;

  v16bf qf[2];
#pragma unroll
  for (int t = 0; t < 2; t++) {
    const __bf16* p = Q + qbase + (size_t)(q0 + r16) * LD + t * 32 + hi * 8;
    FragU u;
    u.u[0] = *(const v4u*)(p);
    u.u[1] = *(const v4u*)(p + 16);
    qf[t] = u.v;
  }

  float mst[8], lst[8];
  v8f o[4];
#pragma unroll
  for (int r = 0; r < 8; r++) { mst[r] = -1e30f; lst[r] = 0.f; }
#pragma unroll
  for (int t = 0; t < 4; t++)
    for (int r = 0; r < 8; r++) o[t][r] = 0.f;

  for (int j0 = 0; j0 < nk; j0 += 32) {
    // stage V chunk transposed: vt[d][k] (zero-pad past nk)
    for (int idx = lane; idx < 256; idx += 32) {
      int jj = idx >> 3;
      int d8 = (idx & 7) * 8;
      H8 dv;
      if (j0 + jj < nk) dv.u = *(const v4u*)(V + kbase + (size_t)(j0 + jj) * LD + d8);
      else { v4u z = {0u,0u,0u,0u}; dv.u = z; }
#pragma unroll
      for (int e = 0; e < 8; e++) vt[(d8 + e) * 32 + jj] = dv.e[e];
    }
    __syncthreads();

    // scores for two 16-key tiles
    v8f sc[2];
#pragma unroll
    for (int t = 0; t < 2; t++) {
      int jc = j0 + t * 16 + r16;
      v16bf kf0, kf1;
      if (jc < nk) {
        const __bf16* p = Kp + kbase + (size_t)jc * LD + hi * 8;
        FragU u0, u1;
        u0.u[0] = *(const v4u*)(p);      u0.u[1] = *(const v4u*)(p + 16);
        u1.u[0] = *(const v4u*)(p + 32); u1.u[1] = *(const v4u*)(p + 48);
        kf0 = u0.v; kf1 = u1.v;
      } else { kf0 = zero_frag(); kf1 = zero_frag(); }
      v8f c;
      for (int r = 0; r < 8; r++) c[r] = 0.f;
      c = WMMA_BF16(qf[0], kf0, c);
      c = WMMA_BF16(qf[1], kf1, c);
#pragma unroll
      for (int r = 0; r < 8; r++) sc[t][r] = (jc < nk) ? c[r] * 0.125f : -1e30f;
    }

    // online softmax (rows striped: elem r -> row r + hi*8, col r16 in lanes)
#pragma unroll
    for (int r = 0; r < 8; r++) {
      float a0 = sc[0][r], a1 = sc[1][r];
      float vm = fmaxf(a0, a1);
      for (int m = 8; m >= 1; m >>= 1) vm = fmaxf(vm, __shfl_xor(vm, m, 16));
      float mnew = fmaxf(mst[r], vm);
      float fct  = __expf(mst[r] - mnew);
      float p0   = __expf(a0 - mnew);
      float p1   = __expf(a1 - mnew);
      float rsum = p0 + p1;
      for (int m = 8; m >= 1; m >>= 1) rsum += __shfl_xor(rsum, m, 16);
      lst[r] = lst[r] * fct + rsum;
      mst[r] = mnew;
#pragma unroll
      for (int t = 0; t < 4; t++) o[t][r] *= fct;
      int prow = r + hi * 8;
      pp[prow * 32 + r16]      = p0;
      pp[prow * 32 + 16 + r16] = p1;
    }
    __syncthreads();

    // rebuild P as WMMA A-fragment (16x32, K = 32 keys)
    v16bf pf;
    {
      const float* pr = pp + r16 * 32 + hi * 8;
      FragU u;
#pragma unroll
      for (int i = 0; i < 8; i++) u.e[i]     = (__bf16)pr[i];
#pragma unroll
      for (int i = 0; i < 8; i++) u.e[8 + i] = (__bf16)pr[16 + i];
      pf = u.v;
    }
#pragma unroll
    for (int t = 0; t < 4; t++) {
      v16bf vf = load_b16(&vt[(t * 16 + r16) * 32 + hi * 8]);
      o[t] = WMMA_BF16(pf, vf, o[t]);
    }
    __syncthreads();
  }

  // normalize and store (bf16, same [b*nq + i][1024] layout for O-projection)
#pragma unroll
  for (int t = 0; t < 4; t++) {
    int col = t * 16 + r16;
#pragma unroll
    for (int r = 0; r < 8; r++) {
      int row = q0 + r + hi * 8;
      float val = o[t][r] / lst[r];
      O[qbase + (size_t)row * LD + col] = (__bf16)val;
    }
  }
}

// ---------------------------------------------------------------------------
// Host orchestration
// ---------------------------------------------------------------------------
extern "C" void kernel_launch(void* const* d_in, const int* in_sizes, int n_in,
                              void* d_out, int out_size, void* d_ws, size_t ws_size,
                              hipStream_t stream) {
  const float* x   = (const float*)d_in[0];
  const float* ctx = (const float*)d_in[1];
  const float* Wq1 = (const float*)d_in[2];
  const float* Wk1 = (const float*)d_in[3];
  const float* Wv1 = (const float*)d_in[4];
  const float* Wo1 = (const float*)d_in[5];
  const float* bo1 = (const float*)d_in[6];
  const float* Wq2 = (const float*)d_in[7];
  const float* Wk2 = (const float*)d_in[8];
  const float* Wv2 = (const float*)d_in[9];
  const float* Wo2 = (const float*)d_in[10];
  const float* bo2 = (const float*)d_in[11];
  const float* Wp  = (const float*)d_in[12];
  const float* bp  = (const float*)d_in[13];
  const float* W2  = (const float*)d_in[14];
  const float* b2  = (const float*)d_in[15];
  const float* g1  = (const float*)d_in[16];
  const float* be1 = (const float*)d_in[17];
  const float* g2  = (const float*)d_in[18];
  const float* be2 = (const float*)d_in[19];
  const float* g3  = (const float*)d_in[20];
  const float* be3 = (const float*)d_in[21];

  const int Bb = 2, Nq = 2048, D = 1024, CT = 768, Mk = 77;
  const int R = Bb * Nq;           // 4096 rows of x
  const int Rk = Bb * Mk;          // 154 context rows

  char* wsp = (char*)d_ws;
  auto alloc = [&](size_t bytes) -> char* {
    char* p = wsp;
    wsp += (bytes + 255) & ~(size_t)255;
    return p;
  };
  __bf16* wq1b = (__bf16*)alloc((size_t)D * D * 2);
  __bf16* wk1b = (__bf16*)alloc((size_t)D * D * 2);
  __bf16* wv1b = (__bf16*)alloc((size_t)D * D * 2);
  __bf16* wo1b = (__bf16*)alloc((size_t)D * D * 2);
  __bf16* wq2b = (__bf16*)alloc((size_t)D * D * 2);
  __bf16* wk2b = (__bf16*)alloc((size_t)CT * D * 2);
  __bf16* wv2b = (__bf16*)alloc((size_t)CT * D * 2);
  __bf16* wo2b = (__bf16*)alloc((size_t)D * D * 2);
  __bf16* wpb  = (__bf16*)alloc((size_t)D * 8192 * 2);
  __bf16* w2b  = (__bf16*)alloc((size_t)4096 * D * 2);
  __bf16* ctxb = (__bf16*)alloc((size_t)Rk * CT * 2);
  __bf16* hb   = (__bf16*)alloc((size_t)R * D * 2);
  __bf16* qb   = (__bf16*)alloc((size_t)R * D * 2);
  __bf16* kb   = (__bf16*)alloc((size_t)R * D * 2);
  __bf16* vb   = (__bf16*)alloc((size_t)R * D * 2);
  __bf16* aob  = (__bf16*)alloc((size_t)R * D * 2);
  __bf16* kcb  = (__bf16*)alloc((size_t)Rk * D * 2);
  __bf16* vcb  = (__bf16*)alloc((size_t)Rk * D * 2);
  __bf16* hidb = (__bf16*)alloc((size_t)R * 4096 * 2);
  float*  x1   = (float*)alloc((size_t)R * D * 4);
  float*  x2   = (float*)alloc((size_t)R * D * 4);

  auto conv = [&](const float* s, __bf16* dst, size_t n) {
    unsigned blocks = (unsigned)((n + 1023) / 1024);
    cvt_bf16_kernel<<<dim3(blocks), dim3(256), 0, stream>>>(s, dst, (long long)n);
  };
  conv(Wq1, wq1b, (size_t)D * D);
  conv(Wk1, wk1b, (size_t)D * D);
  conv(Wv1, wv1b, (size_t)D * D);
  conv(Wo1, wo1b, (size_t)D * D);
  conv(Wq2, wq2b, (size_t)D * D);
  conv(Wk2, wk2b, (size_t)CT * D);
  conv(Wv2, wv2b, (size_t)CT * D);
  conv(Wo2, wo2b, (size_t)D * D);
  conv(Wp,  wpb,  (size_t)D * 8192);
  conv(W2,  w2b,  (size_t)4096 * D);
  conv(ctx, ctxb, (size_t)Rk * CT);

  auto gemm = [&](const __bf16* A, int lda, const __bf16* Bm, int ldb,
                  const __bf16* B2, const float* bias, const float* bias2,
                  const float* res, float* oF, __bf16* oB, int ldo,
                  int M, int N, int Kd) {
    dim3 grid(N / 64, (M + 63) / 64);
    gemm_bf16_kernel<<<grid, dim3(128), 0, stream>>>(
        A, lda, Bm, ldb, B2, bias, bias2, res, oF, oB, ldo, M, N, Kd);
  };

  // ---- block 1: self-attention ----
  ln_bf16_kernel<<<dim3(R), dim3(256), 0, stream>>>(x, g1, be1, hb, D);
  gemm(hb, D, wq1b, D, nullptr, nullptr, nullptr, nullptr, nullptr, qb, D, R, D, D);
  gemm(hb, D, wk1b, D, nullptr, nullptr, nullptr, nullptr, nullptr, kb, D, R, D, D);
  gemm(hb, D, wv1b, D, nullptr, nullptr, nullptr, nullptr, nullptr, vb, D, R, D, D);
  flash_attn_kernel<<<dim3(Nq / 16, 16, Bb), dim3(32), 0, stream>>>(qb, kb, vb, aob, Nq, Nq);
  gemm(aob, D, wo1b, D, nullptr, bo1, nullptr, x, x1, nullptr, D, R, D, D);

  // ---- block 2: cross-attention ----
  ln_bf16_kernel<<<dim3(R), dim3(256), 0, stream>>>(x1, g2, be2, hb, D);
  gemm(hb, D, wq2b, D, nullptr, nullptr, nullptr, nullptr, nullptr, qb, D, R, D, D);
  gemm(ctxb, CT, wk2b, D, nullptr, nullptr, nullptr, nullptr, nullptr, kcb, D, Rk, D, CT);
  gemm(ctxb, CT, wv2b, D, nullptr, nullptr, nullptr, nullptr, nullptr, vcb, D, Rk, D, CT);
  flash_attn_kernel<<<dim3(Nq / 16, 16, Bb), dim3(32), 0, stream>>>(qb, kcb, vcb, aob, Nq, Mk);
  gemm(aob, D, wo2b, D, nullptr, bo2, nullptr, x1, x2, nullptr, D, R, D, D);

  // ---- GEGLU FFN ----
  ln_bf16_kernel<<<dim3(R), dim3(256), 0, stream>>>(x2, g3, be3, hb, D);
  gemm(hb, D, wpb, 8192, wpb + 4096, bp, bp + 4096, nullptr, nullptr, hidb, 4096,
       R, 4096, D);
  gemm(hidb, 4096, w2b, D, nullptr, b2, nullptr, x2, (float*)d_out, nullptr, D,
       R, D, 4096);
}
